// SID_ASPP_79731772883308
// MI455X (gfx1250) — matrix operand order repaired and checked by reading, not compiled, not run on hardware
//
#include <hip/hip_runtime.h>
#include <hip/hip_bf16.h>
#include <math.h>

// ---------------------------------------------------------------------------
// SID-ASPP fused implementation for gfx1250 (MI455X).
// Big GEMMs (conv0 256x256, proj 256x3840) use v_wmma_f32_16x16x32_bf16 with
// 32x32 register tiles per wave (2 A frags x 2 B frags -> 4 WMMAs per K-step)
// to double the WMMA:VMEM issue ratio vs a single 16x16 tile.
// r (B, HW, 15*256) is kept in bf16 channel-last so WMMA B fragments are
// contiguous 32B loads per lane (lane = pixel).
// ---------------------------------------------------------------------------

#define B_  8
#define C_  256
#define H_  48
#define W_  48
#define HW_ 2304
#define RCH 3840          // 15 * 256
#define NBR 12

typedef __bf16 v16bf __attribute__((ext_vector_type(16)));
typedef __bf16 v8bf  __attribute__((ext_vector_type(8)));
typedef __bf16 v4bf  __attribute__((ext_vector_type(4)));
typedef float  v8f   __attribute__((ext_vector_type(8)));

__constant__ int BR_IDX[NBR] = {-1,0,1,-1,0,1,-1,0,1,-1,0,1};
__constant__ int BR_DIL[NBR] = { 1,1,1, 3,3,3, 6,6,6,12,12,12};

// -------------------- small prep kernels --------------------

__global__ void k_bn(const float* __restrict__ g, const float* __restrict__ b,
                     const float* __restrict__ m, const float* __restrict__ v,
                     float* __restrict__ s, float* __restrict__ t, int n) {
  int i = blockIdx.x * blockDim.x + threadIdx.x;
  if (i < n) {
    float inv = g[i] * rsqrtf(v[i] + 1e-5f);
    s[i] = inv;
    t[i] = b[i] - m[i] * inv;
  }
}

__global__ void k_cvt(const float* __restrict__ src, __bf16* __restrict__ dst, int n) {
  int i = blockIdx.x * blockDim.x + threadIdx.x;
  if (i < n) dst[i] = (__bf16)src[i];
}

// -------------------- 3x3 pooled stats + global mean --------------------
// grid (C_, B_), block 288 (9 waves, wave per 16x16 bin)
__global__ void k_pool(const float* __restrict__ x, float* __restrict__ pool33,
                       float* __restrict__ gm) {
  int c = blockIdx.x, b = blockIdx.y;
  int wv = threadIdx.x >> 5, lane = threadIdx.x & 31;
  __shared__ float bins[9];
  const float* xp = x + ((size_t)b * C_ + c) * HW_;
  if (wv < 9) {
    int bh = wv / 3, bw = wv % 3;
    float s = 0.f;
    #pragma unroll
    for (int k = 0; k < 8; k++) {
      int idxp = lane * 8 + k;                     // 0..255 within bin
      int hh = bh * 16 + (idxp >> 4);
      int ww = bw * 16 + (idxp & 15);
      s += xp[hh * W_ + ww];
    }
    for (int m = 16; m >= 1; m >>= 1) s += __shfl_xor(s, m, 32);
    if (lane == 0) bins[wv] = s * (1.f / 256.f);
  }
  __syncthreads();
  if (threadIdx.x == 0) {
    float g = 0.f;
    for (int k = 0; k < 9; k++) { pool33[((size_t)b * C_ + c) * 9 + k] = bins[k]; g += bins[k]; }
    gm[b * C_ + c] = g * (1.f / 9.f);
  }
}

// -------------------- per-branch dynamic weight/bias --------------------

struct BrP { const float *b1, *b2, *bdyn, *w1, *w2, *wdyn; };
struct AllBr { BrP p[NBR]; };

// grid (NBR, B_), block 256
__global__ void k_branch_dyn(AllBr ab,
                             const float* __restrict__ pool33, const float* __restrict__ gm,
                             const float* __restrict__ bn1s, const float* __restrict__ bn1t,
                             float* __restrict__ dynw, float* __restrict__ dynb) {
  int br = blockIdx.x, b = blockIdx.y, t = threadIdx.x;
  int idx = BR_IDX[br];
  int kh = (idx == 1) ? 1 : 3, kw = (idx == 0) ? 1 : 3;
  int taps = kh * kw, cols = taps + 1;             // last col = global mean
  BrP P = ab.p[br];
  __shared__ float zs[10 * 256];
  __shared__ float hb[10 * 64];

  { // stage pooled z columns + gm column (thread = channel)
    int c = t;
    const float* p33 = pool33 + ((size_t)b * C_ + c) * 9;
    if (idx == -1) { for (int k = 0; k < 9; k++) zs[k * 256 + c] = p33[k]; }
    else if (idx == 0) { for (int i = 0; i < 3; i++) zs[i * 256 + c] = (p33[i*3] + p33[i*3+1] + p33[i*3+2]) * (1.f/3.f); }
    else { for (int j = 0; j < 3; j++) zs[j * 256 + c] = (p33[j] + p33[3+j] + p33[6+j]) * (1.f/3.f); }
    zs[taps * 256 + c] = gm[b * C_ + c];
  }
  __syncthreads();

  // h = gelu(bn1(W1 z + b1)), stored column-major
  for (int e = t; e < 64 * cols; e += 256) {
    int row = e % 64, col = e / 64;
    float acc = P.b1[row];
    const float* wr = P.w1 + row * 256;
    const float* zc = zs + col * 256;
    for (int c = 0; c < 256; c++) acc += wr[c] * zc[c];
    float v = acc * bn1s[br * 64 + row] + bn1t[br * 64 + row];
    v = 0.5f * v * (1.f + erff(v * 0.70710678118f));
    hb[col * 64 + row] = v;
  }
  __syncthreads();

  // dynamic tap weights: softmax over G of (W2 h + b2), dot wdyn
  for (int e = t; e < 256 * taps; e += 256) {
    int c = e & 255, tap = e >> 8;
    float sg[4];
    const float* hc = hb + tap * 64;
    #pragma unroll
    for (int g = 0; g < 4; g++) {
      int row = g * 256 + c;
      float acc = P.b2[row];
      const float* wr = P.w2 + row * 64;
      for (int r = 0; r < 64; r++) acc += wr[r] * hc[r];
      sg[g] = acc;
    }
    float mx = fmaxf(fmaxf(sg[0], sg[1]), fmaxf(sg[2], sg[3]));
    float e0 = expf(sg[0]-mx), e1 = expf(sg[1]-mx), e2 = expf(sg[2]-mx), e3 = expf(sg[3]-mx);
    float inv = 1.f / (e0 + e1 + e2 + e3);
    float wvl = (e0 * P.wdyn[(0*256+c)*taps + tap] + e1 * P.wdyn[(1*256+c)*taps + tap] +
                 e2 * P.wdyn[(2*256+c)*taps + tap] + e3 * P.wdyn[(3*256+c)*taps + tap]) * inv;
    dynw[(((size_t)br * B_ + b) * C_ + c) * 9 + tap] = wvl;
  }

  { // dynamic bias from gm column
    int c = t;
    float sg[4];
    const float* hc = hb + taps * 64;
    #pragma unroll
    for (int g = 0; g < 4; g++) {
      int row = g * 256 + c;
      float acc = P.b2[row];
      const float* wr = P.w2 + row * 64;
      for (int r = 0; r < 64; r++) acc += wr[r] * hc[r];
      sg[g] = acc;
    }
    float mx = fmaxf(fmaxf(sg[0], sg[1]), fmaxf(sg[2], sg[3]));
    float e0 = expf(sg[0]-mx), e1 = expf(sg[1]-mx), e2 = expf(sg[2]-mx), e3 = expf(sg[3]-mx);
    float inv = 1.f / (e0 + e1 + e2 + e3);
    float bv = (e0 * P.bdyn[0*256+c] + e1 * P.bdyn[1*256+c] +
                e2 * P.bdyn[2*256+c] + e3 * P.bdyn[3*256+c]) * inv;
    dynb[((size_t)br * B_ + b) * C_ + c] = bv;
  }
}

// -------------------- pool branch (tiny matvec) --------------------
// grid (B_), block 256
__global__ void k_poolbr(const float* __restrict__ pw, const float* __restrict__ gm,
                         const float* __restrict__ s, const float* __restrict__ t,
                         float* __restrict__ yb) {
  int b = blockIdx.x, oc = threadIdx.x;
  const float* g = gm + b * C_;
  const float* wr = pw + oc * C_;
  float acc = 0.f;
  for (int c = 0; c < C_; c++) acc += wr[c] * g[c];
  yb[b * C_ + oc] = fmaxf(acc * s[oc] + t[oc], 0.f);
}

// -------------------- x -> r block0 (bf16, channel-last) + block14 broadcast ----
// grid (HW_/256, B_), block 256 (thread = pixel)
__global__ void k_xpose(const float* __restrict__ x, const float* __restrict__ yb,
                        __bf16* __restrict__ R) {
  int b = blockIdx.y;
  int p = blockIdx.x * 256 + threadIdx.x;
  const float* xb = x + (size_t)b * C_ * HW_;
  __bf16* rp = R + ((size_t)b * HW_ + p) * RCH;
  for (int cb = 0; cb < C_; cb += 8) {
    v8bf pk, pk2;
    #pragma unroll
    for (int k = 0; k < 8; k++) {
      pk[k]  = (__bf16)xb[(size_t)(cb + k) * HW_ + p];
      pk2[k] = (__bf16)yb[b * C_ + cb + k];
    }
    *(v8bf*)(rp + cb)            = pk;   // block 0 = x
    *(v8bf*)(rp + 14 * C_ + cb)  = pk2;  // block 14 = broadcast pool branch
  }
}

// -------------------- WMMA helpers --------------------

__device__ inline v16bf ldA16x32(const __bf16* base) {
  // A fragment per lane: K = {0..15} relative to base (half-offset applied by caller)
  v8bf a0 = *(const v8bf*)(base);
  v8bf a1 = *(const v8bf*)(base + 8);
  return __builtin_shufflevector(a0, a1, 0,1,2,3,4,5,6,7,8,9,10,11,12,13,14,15);
}

#define WMMA_BF16(a, b, c) \
  __builtin_amdgcn_wmma_f32_16x16x32_bf16(false, (a), false, (b), (short)0, (c), false, false)

// -------------------- conv0 GEMM: r_blk1 = relu(bn0(W0 @ x)) --------------------
// grid (72, B_), block 256 (8 waves; wave = one 32x32 register tile, 4 WMMAs/K-step)
__global__ void k_conv0(const __bf16* __restrict__ A /*256x256*/, __bf16* __restrict__ R,
                        const float* __restrict__ s, const float* __restrict__ t) {
  int b = blockIdx.y;
  int wv = threadIdx.x >> 5, lane = threadIdx.x & 31;
  int l16 = lane & 15, hf = lane >> 4;
  int tile = blockIdx.x * 8 + wv;                  // 0..575
  int mT = tile & 7, nT = tile >> 3;               // 8 M-tiles(32), 72 N-tiles(32)
  __bf16* Rb = R + (size_t)b * HW_ * RCH;
  const __bf16* arow0 = A + (size_t)(mT * 32 + l16) * 256;
  const __bf16* arow1 = arow0 + (size_t)16 * 256;
  const __bf16* bcol0 = Rb + (size_t)(nT * 32 + l16) * RCH;   // block0 channels
  const __bf16* bcol1 = bcol0 + (size_t)16 * RCH;
  v8f acc00 = {}, acc01 = {}, acc10 = {}, acc11 = {};
  #pragma unroll 2
  for (int kk = 0; kk < 256; kk += 32) {
    v16bf a0 = ldA16x32(arow0 + kk + hf * 16);
    v16bf a1 = ldA16x32(arow1 + kk + hf * 16);
    v16bf b0 = *(const v16bf*)(bcol0 + kk + hf * 16);
    v16bf b1 = *(const v16bf*)(bcol1 + kk + hf * 16);
    acc00 = WMMA_BF16(a0, b0, acc00);
    acc01 = WMMA_BF16(a0, b1, acc01);
    acc10 = WMMA_BF16(a1, b0, acc10);
    acc11 = WMMA_BF16(a1, b1, acc11);
  }
  int pix0 = nT * 32 + l16, pix1 = pix0 + 16;
  int ocb0 = mT * 32 + hf * 8, ocb1 = ocb0 + 16;
  v8bf pk;
  #pragma unroll
  for (int v = 0; v < 8; v++) pk[v] = (__bf16)fmaxf(acc00[v] * s[ocb0 + v] + t[ocb0 + v], 0.f);
  *(v8bf*)(Rb + (size_t)pix0 * RCH + C_ + ocb0) = pk;
  #pragma unroll
  for (int v = 0; v < 8; v++) pk[v] = (__bf16)fmaxf(acc01[v] * s[ocb0 + v] + t[ocb0 + v], 0.f);
  *(v8bf*)(Rb + (size_t)pix1 * RCH + C_ + ocb0) = pk;
  #pragma unroll
  for (int v = 0; v < 8; v++) pk[v] = (__bf16)fmaxf(acc10[v] * s[ocb1 + v] + t[ocb1 + v], 0.f);
  *(v8bf*)(Rb + (size_t)pix0 * RCH + C_ + ocb1) = pk;
  #pragma unroll
  for (int v = 0; v < 8; v++) pk[v] = (__bf16)fmaxf(acc11[v] * s[ocb1 + v] + t[ocb1 + v], 0.f);
  *(v8bf*)(Rb + (size_t)pix1 * RCH + C_ + ocb1) = pk;
}

// -------------------- dynamic depthwise branches --------------------
// grid (HW_/256, B_, NBR), block 256 (thread = pixel)
__global__ void k_branch_apply(const float* __restrict__ x,
                               const float* __restrict__ dynw, const float* __restrict__ dynb,
                               const float* __restrict__ bos, const float* __restrict__ bot,
                               __bf16* __restrict__ R) {
  int br = blockIdx.z, b = blockIdx.y;
  int p = blockIdx.x * 256 + threadIdx.x;
  int h = p / W_, wq = p % W_;
  int idx = BR_IDX[br], dil = BR_DIL[br];
  int kh = (idx == 1) ? 1 : 3, kw = (idx == 0) ? 1 : 3;
  int ph = (kh == 3) ? ((dil == 1) ? 1 : dil) : 0;
  int pw = (kw == 3) ? ((dil == 1) ? 1 : dil) : 0;
  const float* xb = x + (size_t)b * C_ * HW_;
  __bf16* rp = R + ((size_t)b * HW_ + p) * RCH + (2 + br) * C_;
  for (int cb = 0; cb < C_; cb += 8) {
    v8bf pk;
    #pragma unroll
    for (int cc = 0; cc < 8; cc++) {
      int c = cb + cc;
      const float* wt = dynw + (((size_t)br * B_ + b) * C_ + c) * 9;
      float acc = dynb[((size_t)br * B_ + b) * C_ + c];
      const float* xp = xb + (size_t)c * HW_;
      for (int i = 0; i < kh; i++) {
        int hs = h + i * dil - ph;
        for (int j = 0; j < kw; j++) {
          int wsq = wq + j * dil - pw;
          float xv = (hs >= 0 && hs < H_ && wsq >= 0 && wsq < W_) ? xp[hs * W_ + wsq] : 0.f;
          acc += wt[i * kw + j] * xv;
        }
      }
      pk[cc] = (__bf16)fmaxf(acc * bos[br * C_ + c] + bot[br * C_ + c], 0.f);
    }
    *(v8bf*)(rp + cb) = pk;
  }
}

// -------------------- channel sums of r (for ECA) --------------------
// grid (RCH/64, B_), block 256
__global__ void k_chsum(const __bf16* __restrict__ R, float* __restrict__ chsum) {
  int g = blockIdx.x, b = blockIdx.y;
  int t = threadIdx.x;
  int ch = g * 64 + (t & 63), pr = t >> 6;
  const __bf16* rb = R + (size_t)b * HW_ * RCH;
  float s = 0.f;
  for (int p = pr; p < HW_; p += 4) s += (float)rb[(size_t)p * RCH + ch];
  __shared__ float red[256];
  red[t] = s;
  __syncthreads();
  if (t < 64) chsum[b * RCH + g * 64 + t] = red[t] + red[t + 64] + red[t + 128] + red[t + 192];
}

// -------------------- ECA attention --------------------
// grid (B_*RCH/256), block 256
__global__ void k_eca(const float* __restrict__ chsum, const float* __restrict__ ew,
                      float* __restrict__ att) {
  int i = blockIdx.x * 256 + threadIdx.x;
  int ch = i % RCH;
  const float sc = 1.f / (float)HW_;
  float cm  = chsum[i] * sc;
  float cml = (ch > 0)        ? chsum[i - 1] * sc : 0.f;
  float cmr = (ch < RCH - 1)  ? chsum[i + 1] * sc : 0.f;
  float z = ew[0] * cml + ew[1] * cm + ew[2] * cmr;
  att[i] = 1.f / (1.f + expf(-z));
}

// -------------------- fold att into proj weights (per batch), bf16 --------------------
// grid (B_*C_*RCH/4/256), block 256
__global__ void k_fold(const float* __restrict__ pw, const float* __restrict__ att,
                       __bf16* __restrict__ WA) {
  size_t i = ((size_t)blockIdx.x * 256 + threadIdx.x) * 4;   // over B_*C_*RCH
  int b = (int)(i / ((size_t)C_ * RCH));
  size_t r = i % ((size_t)C_ * RCH);
  int ch = (int)(r % RCH);
  float4 w4 = *(const float4*)(pw + r);
  float4 a4 = *(const float4*)(att + (size_t)b * RCH + ch);
  v4bf o;
  o[0] = (__bf16)(w4.x * a4.x); o[1] = (__bf16)(w4.y * a4.y);
  o[2] = (__bf16)(w4.z * a4.z); o[3] = (__bf16)(w4.w * a4.w);
  *(v4bf*)(WA + i) = o;
}

// -------------------- proj GEMM: out = relu(bn_proj(WA @ r)) + x --------------------
// grid (72, B_), block 256 (8 waves; wave = one 32x32 register tile; K = 3840)
__global__ void k_proj(const __bf16* __restrict__ WA, const __bf16* __restrict__ R,
                       const float* __restrict__ s, const float* __restrict__ t,
                       const float* __restrict__ x, float* __restrict__ out) {
  int b = blockIdx.y;
  int wv = threadIdx.x >> 5, lane = threadIdx.x & 31;
  int l16 = lane & 15, hf = lane >> 4;
  int tile = blockIdx.x * 8 + wv;                  // 0..575
  int mT = tile & 7, nT = tile >> 3;               // 8 M-tiles(32), 72 N-tiles(32)
  const __bf16* arow0 = WA + (size_t)b * C_ * RCH + (size_t)(mT * 32 + l16) * RCH;
  const __bf16* arow1 = arow0 + (size_t)16 * RCH;
  const __bf16* bcol0 = R + ((size_t)b * HW_ + nT * 32 + l16) * RCH;
  const __bf16* bcol1 = bcol0 + (size_t)16 * RCH;
  v8f acc00 = {}, acc01 = {}, acc10 = {}, acc11 = {};
  #pragma unroll 2
  for (int kk = 0; kk < RCH; kk += 32) {
    v16bf a0 = ldA16x32(arow0 + kk + hf * 16);
    v16bf a1 = ldA16x32(arow1 + kk + hf * 16);
    v16bf b0 = *(const v16bf*)(bcol0 + kk + hf * 16);
    v16bf b1 = *(const v16bf*)(bcol1 + kk + hf * 16);
    acc00 = WMMA_BF16(a0, b0, acc00);
    acc01 = WMMA_BF16(a0, b1, acc01);
    acc10 = WMMA_BF16(a1, b0, acc10);
    acc11 = WMMA_BF16(a1, b1, acc11);
  }
  int pix0 = nT * 32 + l16, pix1 = pix0 + 16;
  int ocb0 = mT * 32 + hf * 8, ocb1 = ocb0 + 16;
  const float* xb = x + (size_t)b * C_ * HW_;
  float* ob = out + (size_t)b * C_ * HW_;
  #pragma unroll
  for (int v = 0; v < 8; v++) {
    int oc = ocb0 + v;
    ob[(size_t)oc * HW_ + pix0] = fmaxf(acc00[v] * s[oc] + t[oc], 0.f) + xb[(size_t)oc * HW_ + pix0];
    ob[(size_t)oc * HW_ + pix1] = fmaxf(acc01[v] * s[oc] + t[oc], 0.f) + xb[(size_t)oc * HW_ + pix1];
  }
  #pragma unroll
  for (int v = 0; v < 8; v++) {
    int oc = ocb1 + v;
    ob[(size_t)oc * HW_ + pix0] = fmaxf(acc10[v] * s[oc] + t[oc], 0.f) + xb[(size_t)oc * HW_ + pix0];
    ob[(size_t)oc * HW_ + pix1] = fmaxf(acc11[v] * s[oc] + t[oc], 0.f) + xb[(size_t)oc * HW_ + pix1];
  }
}

// ---------------------------------------------------------------------------
// host launcher
// ---------------------------------------------------------------------------
extern "C" void kernel_launch(void* const* d_in, const int* in_sizes, int n_in,
                              void* d_out, int out_size, void* d_ws, size_t ws_size,
                              hipStream_t stream) {
  (void)in_sizes; (void)n_in; (void)out_size; (void)ws_size;
  // Input order: jax pytree flatten (dict keys sorted alphabetically):
  // 0:x | 1-4:bn0{b,g,m,v} | 5-8:bn_pool | 9-12:bn_proj |
  // 13+14*br: {b1,b2,bdyn, bn1{b,g,m,v}, bn_out{b,g,m,v}, w1,w2,wdyn} |
  // 181:conv0_w 182:eca_w 183:pool_w 184:proj_w
  auto IN = [&](int i) { return (const float*)d_in[i]; };

  char* wsb = (char*)d_ws;
  size_t off = 0;
  auto alloc = [&](size_t bytes) { void* p = wsb + off; off = (off + bytes + 255) & ~(size_t)255; return p; };

  float* pool33 = (float*)alloc((size_t)B_ * C_ * 9 * 4);
  float* gm     = (float*)alloc((size_t)B_ * C_ * 4);
  float* bn0s   = (float*)alloc(C_ * 4);
  float* bn0t   = (float*)alloc(C_ * 4);
  float* bnps   = (float*)alloc(C_ * 4);
  float* bnpt   = (float*)alloc(C_ * 4);
  float* bnjs   = (float*)alloc(C_ * 4);
  float* bnjt   = (float*)alloc(C_ * 4);
  float* bn1s   = (float*)alloc(NBR * 64 * 4);
  float* bn1t   = (float*)alloc(NBR * 64 * 4);
  float* bos    = (float*)alloc(NBR * C_ * 4);
  float* bot    = (float*)alloc(NBR * C_ * 4);
  float* dynw   = (float*)alloc((size_t)NBR * B_ * C_ * 9 * 4);
  float* dynb   = (float*)alloc((size_t)NBR * B_ * C_ * 4);
  float* yb     = (float*)alloc((size_t)B_ * C_ * 4);
  float* chsum  = (float*)alloc((size_t)B_ * RCH * 4);
  float* att    = (float*)alloc((size_t)B_ * RCH * 4);
  __bf16* w0bf  = (__bf16*)alloc((size_t)C_ * C_ * 2);
  __bf16* WA    = (__bf16*)alloc((size_t)B_ * C_ * RCH * 2);
  __bf16* rbuf  = (__bf16*)alloc((size_t)B_ * HW_ * RCH * 2);

  // 1) fold all BN params into scale/shift
  k_bn<<<1, 256, 0, stream>>>(IN(2),  IN(1),  IN(3),  IN(4),  bn0s, bn0t, C_);
  k_bn<<<1, 256, 0, stream>>>(IN(6),  IN(5),  IN(7),  IN(8),  bnps, bnpt, C_);
  k_bn<<<1, 256, 0, stream>>>(IN(10), IN(9),  IN(11), IN(12), bnjs, bnjt, C_);
  for (int br = 0; br < NBR; br++) {
    int bi = 13 + 14 * br;
    k_bn<<<1, 256, 0, stream>>>(IN(bi+4), IN(bi+3), IN(bi+5), IN(bi+6),
                                bn1s + br * 64, bn1t + br * 64, 64);
    k_bn<<<1, 256, 0, stream>>>(IN(bi+8), IN(bi+7), IN(bi+9), IN(bi+10),
                                bos + br * C_, bot + br * C_, C_);
  }
  k_cvt<<<C_ * C_ / 256, 256, 0, stream>>>(IN(181), w0bf, C_ * C_);

  // 2) pooled stats
  k_pool<<<dim3(C_, B_), 288, 0, stream>>>(IN(0), pool33, gm);

  // 3) dynamic branch weights/biases
  AllBr ab;
  for (int br = 0; br < NBR; br++) {
    int bi = 13 + 14 * br;
    ab.p[br] = { IN(bi+0), IN(bi+1), IN(bi+2), IN(bi+11), IN(bi+12), IN(bi+13) };
  }
  k_branch_dyn<<<dim3(NBR, B_), 256, 0, stream>>>(ab, pool33, gm, bn1s, bn1t, dynw, dynb);

  // 4) pool branch + r block0/block14
  k_poolbr<<<B_, 256, 0, stream>>>(IN(183), gm, bnps, bnpt, yb);
  k_xpose<<<dim3(HW_ / 256, B_), 256, 0, stream>>>(IN(0), yb, rbuf);

  // 5) conv0 via WMMA (32x32 tiles)
  k_conv0<<<dim3(72, B_), 256, 0, stream>>>(w0bf, rbuf, bn0s, bn0t);

  // 6) 12 dynamic depthwise branches
  k_branch_apply<<<dim3(HW_ / 256, B_, NBR), 256, 0, stream>>>(IN(0), dynw, dynb, bos, bot, rbuf);

  // 7) ECA: channel sums -> attention -> fold into proj weights
  k_chsum<<<dim3(RCH / 64, B_), 256, 0, stream>>>(rbuf, chsum);
  k_eca<<<(B_ * RCH) / 256, 256, 0, stream>>>(chsum, IN(182), att);
  k_fold<<<(int)(((size_t)B_ * C_ * RCH / 4) / 256), 256, 0, stream>>>(IN(184), att, WA);

  // 8) projection via WMMA (32x32 tiles, + BN + ReLU + residual)
  k_proj<<<dim3(72, B_), 256, 0, stream>>>(WA, rbuf, bnjs, bnjt, IN(0), (float*)d_out);
}
